// BiMamba2_70068096467586
// MI455X (gfx1250) — matrix operand order repaired and checked by reading, not compiled, not run on hardware
//
#include <hip/hip_runtime.h>
#include <hip/hip_bf16.h>
#include <math.h>
#include <stdint.h>

// ---------------------------------------------------------------------------
// BiMamba2 block for MI455X (gfx1250, wave32, WMMA).
// GEMMs: bf16 WMMA (v_wmma_f32_16x16x32_bf16), f32 accumulate.
// Tile staging: async global->LDS (ASYNCcnt) when the toolchain exposes it.
// ---------------------------------------------------------------------------

typedef unsigned short u16;
typedef __attribute__((ext_vector_type(16))) __bf16 v16bf;
typedef __attribute__((ext_vector_type(8)))  float  v8f;
typedef int v4i __attribute__((vector_size(16)));   // matches builtin param type

#define D_MODEL   1024
#define D_INNER   2048
#define D_STATE   64
#define HEADDIM   64
#define NHEADS    32
#define CONV_DIM  2176
#define D_IN_PROJ 4256
#define BATCH     2
#define SEQ       1024
#define ROWS      (BATCH * SEQ)          // 2048

#define AS1 __attribute__((address_space(1)))
#define AS3 __attribute__((address_space(3)))

#if defined(__has_builtin)
#if __has_builtin(__builtin_amdgcn_global_load_async_to_lds_b128) && \
    __has_builtin(__builtin_amdgcn_s_wait_asynccnt)
#define USE_ASYNC_LDS 1
#endif
#endif

__device__ __forceinline__ u16 f2bf(float f) {
    union { float f; unsigned int u; } v; v.f = f;
    unsigned int u = v.u;
    unsigned int r = u + 0x7FFFu + ((u >> 16) & 1u);   // round-to-nearest-even
    return (u16)(r >> 16);
}

__device__ __forceinline__ float silu(float x) { return x / (1.f + expf(-x)); }

// 16-byte global -> LDS stage (async on CDNA5, sync fallback otherwise)
__device__ __forceinline__ void stage16(const u16* __restrict__ src, u16* dst) {
#ifdef USE_ASYNC_LDS
    __builtin_amdgcn_global_load_async_to_lds_b128(
        (AS1 v4i*)(uintptr_t)src,
        (AS3 v4i*)(unsigned)(uintptr_t)dst, 0, 0);
#else
    *(uint4*)dst = *(const uint4*)src;
#endif
}

__device__ __forceinline__ void stage_wait() {
#ifdef USE_ASYNC_LDS
    __builtin_amdgcn_s_wait_asynccnt(0);
#endif
    __syncthreads();
}

// ---------------------------------------------------------------------------
// bf16 WMMA GEMM:  C[M,N] = A[M,K] * W[N,K]^T   (A,W bf16 bits; C f32)
// M % 64 == 0, K % 32 == 0 required; N guarded (weight rows clamped; the
// resulting garbage columns are discarded by the guarded store).
// Workgroup: 256 threads = 8 waves as 2(M) x 4(N); each wave owns a 32x32
// output (2x2 WMMA accumulators) -> 64x128 block tile, 4 WMMA per wave per
// K-step against 8 ds_load_b128.
// ---------------------------------------------------------------------------
__global__ __launch_bounds__(256) void gemm_bf16_wmma(
    const u16* __restrict__ A, const u16* __restrict__ W,
    float* __restrict__ C, int M, int N, int K)
{
    __shared__ u16 lds_a[64][32];    // [m][k]  4 KB
    __shared__ u16 lds_b[128][32];   // [n][k]  8 KB

    const int tid  = threadIdx.x;
    const int lane = tid & 31;
    const int wave = tid >> 5;
    const int wm   = wave >> 2;      // 0..1
    const int wn   = wave & 3;       // 0..3

    const int m0 = blockIdx.y * 64;
    const int n0 = blockIdx.x * 128;

    v8f acc[2][2] = {{{}, {}}, {{}, {}}};

    for (int k0 = 0; k0 < K; k0 += 32) {
        // ---- stage A tile 64x32 (one b128 per thread) ----
        {
            int idx = tid * 8;
            int r = idx >> 5, c = idx & 31;
            const u16* src = A + (size_t)(m0 + r) * K + k0 + c;
            if (k0 + 32 < K) __builtin_prefetch(src + 32, 0, 1);
            stage16(src, &lds_a[r][c]);
        }
        // ---- stage B tile 128x32 (two b128 per thread), row-clamped ----
        #pragma unroll
        for (int i = 0; i < 2; ++i) {
            int idx = (tid + i * 256) * 8;
            int r = idx >> 5, c = idx & 31;
            int nrow = n0 + r; if (nrow > N - 1) nrow = N - 1;
            stage16(W + (size_t)nrow * K + k0 + c, &lds_b[r][c]);
        }
        stage_wait();

        // ---- fragments per CDNA5 VGPR layouts ----
        union Frag { v16bf v; uint4 q[2]; };
        Frag af[2], bf[2];
        {
            int kh = (lane >> 4) * 8;    // A: K 0-7/16-23 vs 8-15/24-31 per lane half
            int kb = (lane >> 4) * 8;    // B: K 0-7/16-23 vs 8-15/24-31 per lane half
            #pragma unroll
            for (int mi = 0; mi < 2; ++mi) {
                int m = wm * 32 + mi * 16 + (lane & 15);
                af[mi].q[0] = *(const uint4*)&lds_a[m][kh];
                af[mi].q[1] = *(const uint4*)&lds_a[m][16 + kh];
            }
            #pragma unroll
            for (int ni = 0; ni < 2; ++ni) {
                int n = wn * 32 + ni * 16 + (lane & 15);
                bf[ni].q[0] = *(const uint4*)&lds_b[n][kb];
                bf[ni].q[1] = *(const uint4*)&lds_b[n][kb + 16];
            }
        }
        #pragma unroll
        for (int mi = 0; mi < 2; ++mi)
            #pragma unroll
            for (int ni = 0; ni < 2; ++ni)
                acc[mi][ni] = __builtin_amdgcn_wmma_f32_16x16x32_bf16(
                    false, af[mi].v, false, bf[ni].v, (short)0,
                    acc[mi][ni], false, false);
        __syncthreads();
    }

    // ---- store: lane holds N=lane%16, rows M = v + 8*(lane>=16) ----
    #pragma unroll
    for (int ni = 0; ni < 2; ++ni) {
        int ncol = n0 + wn * 32 + ni * 16 + (lane & 15);
        if (ncol < N) {
            #pragma unroll
            for (int mi = 0; mi < 2; ++mi) {
                int rbase = m0 + wm * 32 + mi * 16 + (lane >> 4) * 8;
                #pragma unroll
                for (int v = 0; v < 8; ++v)
                    C[(size_t)(rbase + v) * N + ncol] = acc[mi][ni][v];
            }
        }
    }
}

// ---------------------------------------------------------------------------
// f32 -> bf16 conversion kernels
// ---------------------------------------------------------------------------
__global__ void cvt_bf16_kernel(const float* __restrict__ in, u16* __restrict__ out, int n) {
    int i = blockIdx.x * 256 + threadIdx.x;
    if (i < n) out[i] = f2bf(in[i]);
}

// x (B,L,D) -> bf16 with L flipped (for the backward direction)
__global__ void flip_cvt_kernel(const float* __restrict__ x, u16* __restrict__ out) {
    int i = blockIdx.x * 256 + threadIdx.x;           // exactly 2*1024*1024 threads
    int b = i >> 20, rem = i & 1048575;
    int l = rem >> 10, d = rem & 1023;
    out[(((size_t)b << 10) + (1023 - l)) * D_MODEL + d] = f2bf(x[i]);
}

// ---------------------------------------------------------------------------
// Depthwise causal conv (width 4) + SiLU over the xBC slice of zxbcdt
// ---------------------------------------------------------------------------
__global__ void conv_silu_kernel(const float* __restrict__ zx,
                                 const float* __restrict__ cw,
                                 const float* __restrict__ cb,
                                 float* __restrict__ out)
{
    int i = blockIdx.x * 256 + threadIdx.x;
    if (i >= ROWS * CONV_DIM) return;
    int row = i / CONV_DIM, c = i % CONV_DIM;
    int b = row >> 10, l = row & 1023;
    float acc = cb[c];
    #pragma unroll
    for (int k = 0; k < 4; ++k) {
        int l2 = l + k - 3;
        if (l2 >= 0)
            acc += cw[k * CONV_DIM + c] *
                   zx[(size_t)((b << 10) + l2) * D_IN_PROJ + D_INNER + c];
    }
    out[(size_t)row * CONV_DIM + c] = silu(acc);
}

// dt = softplus(dt_raw + bias); dA = exp(dt * -exp(A_log))
__global__ void dtda_kernel(const float* __restrict__ zx,
                            const float* __restrict__ dtb,
                            const float* __restrict__ Alog,
                            float* __restrict__ dt, float* __restrict__ dA)
{
    int i = blockIdx.x * 256 + threadIdx.x;
    if (i >= ROWS * NHEADS) return;
    int row = i >> 5, h = i & 31;
    float r  = zx[(size_t)row * D_IN_PROJ + (D_INNER + CONV_DIM) + h] + dtb[h];
    float sp = (r > 20.f) ? r : log1pf(expf(r));
    dt[i] = sp;
    dA[i] = expf(-expf(Alog[h]) * sp);
}

// ---------------------------------------------------------------------------
// Sequential SSM scan. One block per (b, head): 64 blocks, 256 threads.
// State h[64p][64n] in registers: thread t owns p = t/4, n = (t%4)*16..+15.
// Per-step n-reduction over the 4 sharing lanes via shfl_xor (intra-wave).
// ---------------------------------------------------------------------------
__global__ __launch_bounds__(256) void scan_kernel(
    const float* __restrict__ xconv,   // [ROWS x 2176] (xs | B | C)
    const float* __restrict__ dt,      // [ROWS x 32]
    const float* __restrict__ dA,      // [ROWS x 32]
    const float* __restrict__ Dp,      // [32]
    float* __restrict__ y)             // [ROWS x 2048]
{
    const int bh = blockIdx.x;
    const int b  = bh >> 5;
    const int h  = bh & 31;
    const int tid = threadIdx.x;
    const int p  = tid >> 2;
    const int g  = tid & 3;

    float hstate[16];
    #pragma unroll
    for (int j = 0; j < 16; ++j) hstate[j] = 0.f;

    __shared__ float s_x[64], s_B[64], s_C[64];
    const float Dh = Dp[h];

    for (int l = 0; l < SEQ; ++l) {
        size_t row = ((size_t)b << 10) + l;
        if (tid < 64)        s_x[tid]       = xconv[row * CONV_DIM + h * HEADDIM + tid];
        else if (tid < 128)  s_B[tid - 64]  = xconv[row * CONV_DIM + D_INNER + (tid - 64)];
        else if (tid < 192)  s_C[tid - 128] = xconv[row * CONV_DIM + D_INNER + D_STATE + (tid - 128)];
        __syncthreads();

        float dAv = dA[row * NHEADS + h];
        float dtv = dt[row * NHEADS + h];
        float xp  = s_x[p];
        float coef = dtv * xp;
        float acc = 0.f;
        #pragma unroll
        for (int j = 0; j < 16; ++j) {
            int n = g * 16 + j;
            hstate[j] = fmaf(hstate[j], dAv, coef * s_B[n]);
            acc = fmaf(hstate[j], s_C[n], acc);
        }
        acc += __shfl_xor(acc, 1, 32);
        acc += __shfl_xor(acc, 2, 32);
        if (g == 0) y[row * D_INNER + h * HEADDIM + p] = acc + Dh * xp;
        __syncthreads();
    }
}

// ---------------------------------------------------------------------------
// y = y * silu(z); RMS-norm over 2048; scale by rms_w; emit bf16.
// One block per row.
// ---------------------------------------------------------------------------
__global__ __launch_bounds__(256) void gated_rmsnorm_kernel(
    const float* __restrict__ y, const float* __restrict__ zx,
    const float* __restrict__ rmsw, u16* __restrict__ out)
{
    int row = blockIdx.x;
    __shared__ float red[8];
    float local[8];
    float ss = 0.f;
    #pragma unroll
    for (int i = 0; i < 8; ++i) {
        int c = threadIdx.x + i * 256;
        float z = zx[(size_t)row * D_IN_PROJ + c];
        float v = y[(size_t)row * D_INNER + c] * silu(z);
        local[i] = v;
        ss += v * v;
    }
    #pragma unroll
    for (int o = 16; o > 0; o >>= 1) ss += __shfl_xor(ss, o, 32);
    int w = threadIdx.x >> 5, ln = threadIdx.x & 31;
    if (ln == 0) red[w] = ss;
    __syncthreads();
    if (threadIdx.x == 0) {
        float tot = 0.f;
        for (int i = 0; i < 8; ++i) tot += red[i];
        red[0] = rsqrtf(tot / (float)D_INNER + 1e-5f);
    }
    __syncthreads();
    float scale = red[0];
    #pragma unroll
    for (int i = 0; i < 8; ++i) {
        int c = threadIdx.x + i * 256;
        out[(size_t)row * D_INNER + c] = f2bf(local[i] * scale * rmsw[c]);
    }
}

// combined[b,l,:] = concat(out_fwd[b,l,:], out_bwd[b,L-1-l,:]) as bf16
__global__ void pack_kernel(const float* __restrict__ outf,
                            const float* __restrict__ outb,
                            u16* __restrict__ comb)
{
    int i = blockIdx.x * 256 + threadIdx.x;           // exactly 2048*2048
    int row = i >> 11, j = i & 2047;
    int b = row >> 10, l = row & 1023;
    float v;
    if (j < D_MODEL) v = outf[(size_t)row * D_MODEL + j];
    else             v = outb[(((size_t)b << 10) + (1023 - l)) * D_MODEL + (j - D_MODEL)];
    comb[i] = f2bf(v);
}

// h = x + combined@Wp^T + pb; LayerNorm. One block per row.
__global__ __launch_bounds__(256) void layernorm_kernel(
    const float* __restrict__ x, const float* __restrict__ t,
    const float* __restrict__ pb, const float* __restrict__ lw,
    const float* __restrict__ lb, float* __restrict__ out)
{
    int row = blockIdx.x;
    __shared__ float s1[8], s2[8];
    float vals[4];
    float s = 0.f, sq = 0.f;
    #pragma unroll
    for (int i = 0; i < 4; ++i) {
        int c = threadIdx.x + i * 256;
        float v = x[(size_t)row * D_MODEL + c] + t[(size_t)row * D_MODEL + c] + pb[c];
        vals[i] = v; s += v; sq += v * v;
    }
    #pragma unroll
    for (int o = 16; o > 0; o >>= 1) { s += __shfl_xor(s, o, 32); sq += __shfl_xor(sq, o, 32); }
    int w = threadIdx.x >> 5, ln = threadIdx.x & 31;
    if (ln == 0) { s1[w] = s; s2[w] = sq; }
    __syncthreads();
    if (threadIdx.x == 0) {
        float ts = 0.f, tq = 0.f;
        for (int i = 0; i < 8; ++i) { ts += s1[i]; tq += s2[i]; }
        float mu = ts / (float)D_MODEL;
        float var = tq / (float)D_MODEL - mu * mu;
        s1[0] = mu; s2[0] = rsqrtf(var + 1e-5f);
    }
    __syncthreads();
    float mu = s1[0], inv = s2[0];
    #pragma unroll
    for (int i = 0; i < 4; ++i) {
        int c = threadIdx.x + i * 256;
        out[(size_t)row * D_MODEL + c] = (vals[i] - mu) * inv * lw[c] + lb[c];
    }
}

// ---------------------------------------------------------------------------
// Host orchestration
// ---------------------------------------------------------------------------
extern "C" void kernel_launch(void* const* d_in, const int* in_sizes, int n_in,
                              void* d_out, int out_size, void* d_ws, size_t ws_size,
                              hipStream_t stream)
{
    const float* x        = (const float*)d_in[0];
    const float* in_w[2]  = { (const float*)d_in[1],  (const float*)d_in[9]  };
    const float* conv_w[2]= { (const float*)d_in[2],  (const float*)d_in[10] };
    const float* conv_bs[2]={ (const float*)d_in[3],  (const float*)d_in[11] };
    const float* dt_bias[2]={ (const float*)d_in[4],  (const float*)d_in[12] };
    const float* A_log[2] = { (const float*)d_in[5],  (const float*)d_in[13] };
    const float* D_par[2] = { (const float*)d_in[6],  (const float*)d_in[14] };
    const float* rms_w[2] = { (const float*)d_in[7],  (const float*)d_in[15] };
    const float* out_w[2] = { (const float*)d_in[8],  (const float*)d_in[16] };
    const float* proj_w   = (const float*)d_in[17];
    const float* proj_b   = (const float*)d_in[18];
    const float* ln_w     = (const float*)d_in[19];
    const float* ln_b     = (const float*)d_in[20];
    float* out = (float*)d_out;

    // ---- workspace carve-out ----
    char* ws = (char*)d_ws;
    size_t off = 0;
    auto alloc = [&](size_t bytes) -> void* {
        void* p = ws + off;
        off = (off + bytes + 255) & ~(size_t)255;
        return p;
    };
    u16*   x_bf[2];   x_bf[0]  = (u16*)alloc((size_t)ROWS * D_MODEL * 2);
                      x_bf[1]  = (u16*)alloc((size_t)ROWS * D_MODEL * 2);
    u16*   win_bf[2]; win_bf[0]= (u16*)alloc((size_t)D_IN_PROJ * D_MODEL * 2);
                      win_bf[1]= (u16*)alloc((size_t)D_IN_PROJ * D_MODEL * 2);
    u16*   wout_bf[2];wout_bf[0]=(u16*)alloc((size_t)D_MODEL * D_INNER * 2);
                      wout_bf[1]=(u16*)alloc((size_t)D_MODEL * D_INNER * 2);
    u16*   wproj_bf  = (u16*)alloc((size_t)D_MODEL * D_INNER * 2);
    float* zx[2];     zx[0]    = (float*)alloc((size_t)ROWS * D_IN_PROJ * 4);
                      zx[1]    = (float*)alloc((size_t)ROWS * D_IN_PROJ * 4);
    float* xconv[2];  xconv[0] = (float*)alloc((size_t)ROWS * CONV_DIM * 4);
                      xconv[1] = (float*)alloc((size_t)ROWS * CONV_DIM * 4);
    float* dtb[2];    dtb[0]   = (float*)alloc((size_t)ROWS * NHEADS * 4);
                      dtb[1]   = (float*)alloc((size_t)ROWS * NHEADS * 4);
    float* dAb[2];    dAb[0]   = (float*)alloc((size_t)ROWS * NHEADS * 4);
                      dAb[1]   = (float*)alloc((size_t)ROWS * NHEADS * 4);
    float* yscan[2];  yscan[0] = (float*)alloc((size_t)ROWS * D_INNER * 4);
                      yscan[1] = (float*)alloc((size_t)ROWS * D_INNER * 4);
    u16*   ynorm[2];  ynorm[0] = (u16*)alloc((size_t)ROWS * D_INNER * 2);
                      ynorm[1] = (u16*)alloc((size_t)ROWS * D_INNER * 2);
    float* outdir[2]; outdir[0]= (float*)alloc((size_t)ROWS * D_MODEL * 4);
                      outdir[1]= (float*)alloc((size_t)ROWS * D_MODEL * 4);
    u16*   comb_bf   = (u16*)alloc((size_t)ROWS * D_INNER * 2);
    float* tmp       = (float*)alloc((size_t)ROWS * D_MODEL * 4);

    auto blocks = [](size_t n) { return (unsigned)((n + 255) / 256); };

    // ---- 1. activations & weights -> bf16 ----
    cvt_bf16_kernel<<<blocks((size_t)ROWS * D_MODEL), 256, 0, stream>>>(x, x_bf[0], ROWS * D_MODEL);
    flip_cvt_kernel<<<blocks((size_t)ROWS * D_MODEL), 256, 0, stream>>>(x, x_bf[1]);
    for (int d = 0; d < 2; ++d) {
        cvt_bf16_kernel<<<blocks((size_t)D_IN_PROJ * D_MODEL), 256, 0, stream>>>(
            in_w[d], win_bf[d], D_IN_PROJ * D_MODEL);
        cvt_bf16_kernel<<<blocks((size_t)D_MODEL * D_INNER), 256, 0, stream>>>(
            out_w[d], wout_bf[d], D_MODEL * D_INNER);
    }
    cvt_bf16_kernel<<<blocks((size_t)D_MODEL * D_INNER), 256, 0, stream>>>(
        proj_w, wproj_bf, D_MODEL * D_INNER);

    // ---- 2..8. per-direction pipeline ----
    for (int d = 0; d < 2; ++d) {
        dim3 g_in((D_IN_PROJ + 127) / 128, ROWS / 64);
        gemm_bf16_wmma<<<g_in, 256, 0, stream>>>(x_bf[d], win_bf[d], zx[d],
                                                 ROWS, D_IN_PROJ, D_MODEL);
        conv_silu_kernel<<<blocks((size_t)ROWS * CONV_DIM), 256, 0, stream>>>(
            zx[d], conv_w[d], conv_bs[d], xconv[d]);
        dtda_kernel<<<blocks((size_t)ROWS * NHEADS), 256, 0, stream>>>(
            zx[d], dt_bias[d], A_log[d], dtb[d], dAb[d]);
        scan_kernel<<<BATCH * NHEADS, 256, 0, stream>>>(
            xconv[d], dtb[d], dAb[d], D_par[d], yscan[d]);
        gated_rmsnorm_kernel<<<ROWS, 256, 0, stream>>>(
            yscan[d], zx[d], rms_w[d], ynorm[d]);
        dim3 g_out((D_MODEL + 127) / 128, ROWS / 64);
        gemm_bf16_wmma<<<g_out, 256, 0, stream>>>(ynorm[d], wout_bf[d], outdir[d],
                                                  ROWS, D_MODEL, D_INNER);
    }

    // ---- 9. concat (+flip bwd) -> bf16 ----
    pack_kernel<<<blocks((size_t)ROWS * D_INNER), 256, 0, stream>>>(
        outdir[0], outdir[1], comb_bf);

    // ---- 10. final projection + residual LayerNorm ----
    dim3 g_fin((D_MODEL + 127) / 128, ROWS / 64);
    gemm_bf16_wmma<<<g_fin, 256, 0, stream>>>(comb_bf, wproj_bf, tmp,
                                              ROWS, D_MODEL, D_INNER);
    layernorm_kernel<<<ROWS, 256, 0, stream>>>(x, tmp, proj_b, ln_w, ln_b, out);
}
